// ManifoldCrossAttention_88115549045341
// MI455X (gfx1250) — matrix operand order repaired and checked by reading, not compiled, not run on hardware
//
#include <hip/hip_runtime.h>

#define F_ 3
#define B_ 4
#define C_ 512
#define N_ 576
#define H_ 8
#define D_ 64
#define KV_ (F_ * N_)      // 1728
#define NT16 (N_ / 16)     // 36 token tiles of 16
#define CKT (C_ / 32)      // 16 contraction tiles of 32
#define MT16 (C_ / 16)     // 32 row tiles of 16
#define KVT16 (KV_ / 16)   // 108
#define KVK (KV_ / 32)     // 54
#define FRAG 512           // elements per 16x32 / 32x16 fragment tile

typedef __bf16 bf16;
typedef __attribute__((ext_vector_type(16))) __bf16 v16bf;
typedef __attribute__((ext_vector_type(8)))  float  v8f;

// ---------------------------------------------------------------------------
// Fragment-tiled storage (contiguous k order: k = 16*(lane/16) + e), stored as frag[lane][e]
// so a consumer lane loads its 16 bf16 operands with two b128 loads.
//
// A-fragment (16 rows m x 32 k): lane = m + 16*g, element e for k where
//   g = (k>>4)&1,  e = k&15
// B-fragment (32 k x 16 cols n): lane = n + 16*((k>>4)&1), e = k&15
static __device__ __forceinline__ int a_pos(int m, int k) {
  return ((m & 15) + 16 * ((k >> 4) & 1)) * 16 + (k & 15);
}
static __device__ __forceinline__ int b_pos(int k, int n) {
  return ((n & 15) + 16 * ((k >> 4) & 1)) * 16 + (k & 15);
}

static __device__ __forceinline__ v16bf load_frag(const bf16* frag, int lane) {
  return ((const v16bf*)frag)[lane];  // 32B/lane, 32B aligned (1KB tiles)
}

static __device__ __forceinline__ v8f wmma_bf16(v16bf a, v16bf b, v8f c) {
  return __builtin_amdgcn_wmma_f32_16x16x32_bf16(false, a, false, b, (short)0, c,
                                                 false, false);
}

static __device__ __forceinline__ v8f vzero8() {
  v8f z = {0.f, 0.f, 0.f, 0.f, 0.f, 0.f, 0.f, 0.f};
  return z;
}

// ---------------------------------------------------------------------------
// f32 -> bf16 conversion into fragment-tiled workspace.
// Xf: [fb(12)][kt(16)][nt16(36)][FRAG]  (B-fragments of the feature maps)
// Wf: [p(4)][mt16(32)][kt(16)][FRAG]    (A-fragments of the weights)
__global__ __launch_bounds__(256) void convert_kernel(
    const float* __restrict__ f0, const float* __restrict__ f1,
    const float* __restrict__ f2, const float* __restrict__ Wq,
    const float* __restrict__ Wk, const float* __restrict__ Wv,
    const float* __restrict__ Wo, bf16* __restrict__ Xf,
    bf16* __restrict__ Wf) {
  const int stride = gridDim.x * blockDim.x;
  const int tid = blockIdx.x * blockDim.x + threadIdx.x;

  const int XTOT = F_ * B_ * CKT * NT16 * FRAG;
  for (int i = tid; i < XTOT; i += stride) {
    int pos = i & (FRAG - 1);
    int rem = i >> 9;  // /FRAG
    int nt = rem % NT16;
    rem /= NT16;
    int kt = rem % CKT;
    int fb = rem / CKT;
    int lane = pos >> 4, e = pos & 15;
    int n = nt * 16 + (lane & 15);
    int c = kt * 32 + (lane >> 4) * 16 + e;  // inverse of b_pos
    int f = fb >> 2, b = fb & 3;
    const float* s = (f == 0) ? f0 : (f == 1) ? f1 : f2;
    Xf[i] = (bf16)s[((size_t)b * C_ + c) * N_ + n];
  }

  const int WTOT = 4 * MT16 * CKT * FRAG;
  for (int i = tid; i < WTOT; i += stride) {
    int pos = i & (FRAG - 1);
    int rem = i >> 9;
    int kt = rem % CKT;
    rem /= CKT;
    int mt = rem % MT16;
    int p = rem / MT16;
    int lane = pos >> 4, e = pos & 15;
    int m = mt * 16 + (lane & 15);
    int c = kt * 32 + (lane >> 4) * 16 + e;  // inv a_pos
    const float* s = (p == 0) ? Wq : (p == 1) ? Wk : (p == 2) ? Wv : Wo;
    Wf[i] = (bf16)s[(size_t)m * C_ + c];
  }
}

// ---------------------------------------------------------------------------
// Projection GEMM, LDS-free: operands stream as pre-swizzled fragments.
// grid = (C/64, N/64, 3*F*B); 4 waves, each owns a 16x64 output strip.
// Epilogue scatters Q/K/V into the attention fragment layouts:
//   Qf: [fb][h][qt16(36)][ks(2)][FRAG]   A-frags, rows=tokens, k=dd
//   Kf: [b][h][kvt16(108)][ks(2)][FRAG]  B-frags, k=dd, cols=kv tokens
//   Vf: [b][h][kvk(54)][ddt(4)][FRAG]    B-frags, k=kv tokens, cols=dd
__global__ __launch_bounds__(128) void proj_gemm_kernel(
    const bf16* __restrict__ Wf, const bf16* __restrict__ Xf,
    const float* __restrict__ bq, const float* __restrict__ bk,
    const float* __restrict__ bv, bf16* __restrict__ Qf, bf16* __restrict__ Kf,
    bf16* __restrict__ Vf) {
  const int t = threadIdx.x, lane = t & 31, w = t >> 5;
  const int mb = blockIdx.x * 64, nb = blockIdx.y * 64;
  const int z = blockIdx.z;
  const int p = z / (F_ * B_);
  const int fb = z % (F_ * B_);
  const int f = fb >> 2, b = fb & 3;

  const bf16* Af = Wf + ((size_t)p * MT16 + (mb >> 4) + w) * CKT * FRAG;
  const bf16* Bf = Xf + ((size_t)fb * CKT * NT16 + (nb >> 4)) * FRAG;

  v8f acc[4];
#pragma unroll
  for (int i = 0; i < 4; ++i) acc[i] = vzero8();

  for (int kk = 0; kk < CKT; ++kk) {
    v16bf af = load_frag(Af + (size_t)kk * FRAG, lane);
    if (kk + 1 < CKT) {
      __builtin_prefetch(Af + (size_t)(kk + 1) * FRAG + lane * 16, 0, 0);
      __builtin_prefetch(Bf + (size_t)(kk + 1) * NT16 * FRAG + lane * 16, 0, 0);
    }
#pragma unroll
    for (int ct = 0; ct < 4; ++ct) {
      v16bf bfr = load_frag(Bf + ((size_t)kk * NT16 + ct) * FRAG, lane);
      acc[ct] = wmma_bf16(af, bfr, acc[ct]);
    }
  }

  const float* bias = (p == 0) ? bq : (p == 1) ? bk : bv;
  const float qscale = 0.125f * 1.44269504088896f;  // d^-1/2 * log2(e)
  const int ksel = lane >> 4, ln = lane & 15;
#pragma unroll
  for (int ct = 0; ct < 4; ++ct) {
#pragma unroll
    for (int r = 0; r < 8; ++r) {
      int o = mb + 16 * w + r + 8 * ksel;  // output channel
      int n = nb + ct * 16 + ln;           // token
      float val = acc[ct][r] + bias[o];
      int hh = o >> 6, dd = o & 63;
      if (p == 0) {
        size_t idx = ((((size_t)fb * H_ + hh) * NT16 + (n >> 4)) * 2 + (dd >> 5)) *
                         FRAG +
                     a_pos(n, dd & 31);
        Qf[idx] = (bf16)(val * qscale);
      } else {
        int kv = f * N_ + n;
        if (p == 1) {
          size_t idx =
              ((((size_t)b * H_ + hh) * KVT16 + (kv >> 4)) * 2 + (dd >> 5)) * FRAG +
              b_pos(dd & 31, kv);
          Kf[idx] = (bf16)val;
        } else {
          size_t idx =
              ((((size_t)b * H_ + hh) * KVK + (kv >> 5)) * 4 + (dd >> 4)) * FRAG +
              b_pos(kv & 31, dd);
          Vf[idx] = (bf16)val;
        }
      }
    }
  }
}

// ---------------------------------------------------------------------------
// Flash attention, barrier-free: Q/K/V fragments stream straight from global;
// only the P (C-layout -> A-layout) conversion touches LDS, per-wave private.
// grid = (N/64=9, H=8, F*B=12); 4 waves x 16 query rows.
__global__ __launch_bounds__(128) void attn_kernel(const bf16* __restrict__ Qf,
                                                   const bf16* __restrict__ Kf,
                                                   const bf16* __restrict__ Vf,
                                                   bf16* __restrict__ Of) {
  __shared__ __align__(16) bf16 sP[4][2 * FRAG];  // per-wave P scratch (2KB)

  const int t = threadIdx.x, lane = t & 31, w = t >> 5;
  const int qb = blockIdx.x * 64;
  const int hh = blockIdx.y;
  const int fb = blockIdx.z;
  const int b = fb & 3;
  const int ksel = lane >> 4, ln = lane & 15;

  const bf16* Qbase =
      Qf + (((size_t)fb * H_ + hh) * NT16 + (qb >> 4) + w) * 2 * FRAG;
  const bf16* Kbase = Kf + ((size_t)b * H_ + hh) * KVT16 * 2 * FRAG;
  const bf16* Vbase = Vf + ((size_t)b * H_ + hh) * KVK * 4 * FRAG;

  v16bf aq0 = load_frag(Qbase, lane);
  v16bf aq1 = load_frag(Qbase + FRAG, lane);

  float mrow[8], lrow[8];
  v8f oacc[4];
#pragma unroll
  for (int r = 0; r < 8; ++r) {
    mrow[r] = -1.0e30f;
    lrow[r] = 0.0f;
  }
#pragma unroll
  for (int i = 0; i < 4; ++i) oacc[i] = vzero8();

  for (int kt = 0; kt < KV_ / 64; ++kt) {
    // S = Q K^T (16 x 64), contraction over d=64 as two WMMA k-steps
    v8f s[4];
#pragma unroll
    for (int ct = 0; ct < 4; ++ct) {
      const bf16* kb = Kbase + ((size_t)(kt * 4 + ct) * 2) * FRAG;
      s[ct] = wmma_bf16(aq0, load_frag(kb, lane), vzero8());
      s[ct] = wmma_bf16(aq1, load_frag(kb + FRAG, lane), s[ct]);
    }

    // online softmax, base-2 (scale*log2e folded into Q)
    float alpha[8];
#pragma unroll
    for (int r = 0; r < 8; ++r) {
      float tmax = fmaxf(fmaxf(s[0][r], s[1][r]), fmaxf(s[2][r], s[3][r]));
#pragma unroll
      for (int msk = 8; msk >= 1; msk >>= 1)
        tmax = fmaxf(tmax, __shfl_xor(tmax, msk, 32));
      float mn = fmaxf(mrow[r], tmax);
      alpha[r] = exp2f(mrow[r] - mn);
      mrow[r] = mn;
      float psum = 0.0f;
#pragma unroll
      for (int ct = 0; ct < 4; ++ct) {
        float pv = exp2f(s[ct][r] - mn);
        s[ct][r] = pv;
        psum += pv;
      }
#pragma unroll
      for (int msk = 8; msk >= 1; msk >>= 1) psum += __shfl_xor(psum, msk, 32);
      lrow[r] = lrow[r] * alpha[r] + psum;
    }

    // spill P into A-fragment order (wave-private LDS), rescale O accumulators
#pragma unroll
    for (int ct = 0; ct < 4; ++ct)
#pragma unroll
      for (int r = 0; r < 8; ++r) {
        int row = r + 8 * ksel;
        int col = ct * 16 + ln;
        sP[w][(col >> 5) * FRAG + a_pos(row, col & 31)] = (bf16)s[ct][r];
      }
#pragma unroll
    for (int ct = 0; ct < 4; ++ct)
#pragma unroll
      for (int r = 0; r < 8; ++r) oacc[ct][r] *= alpha[r];

    // O += P * V (contraction over 64 kv)
    v16bf ap0 = load_frag(&sP[w][0], lane);
    v16bf ap1 = load_frag(&sP[w][FRAG], lane);
#pragma unroll
    for (int ddt = 0; ddt < 4; ++ddt) {
      const bf16* vb = Vbase + ((size_t)(2 * kt) * 4 + ddt) * FRAG;
      oacc[ddt] = wmma_bf16(ap0, load_frag(vb, lane), oacc[ddt]);
      oacc[ddt] = wmma_bf16(ap1, load_frag(vb + 4 * FRAG, lane), oacc[ddt]);
    }
  }

  // write O as B-fragments for the output GEMM: Of [fb][kt(16)][nt16(36)][FRAG]
#pragma unroll
  for (int ddt = 0; ddt < 4; ++ddt)
#pragma unroll
    for (int r = 0; r < 8; ++r) {
      int n = qb + 16 * w + r + 8 * ksel;
      int c = hh * D_ + ddt * 16 + ln;
      float val = oacc[ddt][r] / lrow[r];
      size_t idx = (((size_t)fb * CKT + (c >> 5)) * NT16 + (n >> 4)) * FRAG +
                   b_pos(c & 31, n);
      Of[idx] = (bf16)val;
    }
}

// ---------------------------------------------------------------------------
// Output projection + residual (LDS-free): out = feat + Wo * O + bo (f32)
__global__ __launch_bounds__(128) void out_gemm_kernel(
    const bf16* __restrict__ Wf, const bf16* __restrict__ Of,
    const float* __restrict__ bo, const float* __restrict__ f0,
    const float* __restrict__ f1, const float* __restrict__ f2,
    float* __restrict__ out) {
  const int t = threadIdx.x, lane = t & 31, w = t >> 5;
  const int mb = blockIdx.x * 64, nb = blockIdx.y * 64;
  const int fb = blockIdx.z;
  const int f = fb >> 2, b = fb & 3;

  const bf16* Af = Wf + ((size_t)3 * MT16 + (mb >> 4) + w) * CKT * FRAG;
  const bf16* Bf = Of + ((size_t)fb * CKT * NT16 + (nb >> 4)) * FRAG;

  v8f acc[4];
#pragma unroll
  for (int i = 0; i < 4; ++i) acc[i] = vzero8();

  for (int kk = 0; kk < CKT; ++kk) {
    v16bf af = load_frag(Af + (size_t)kk * FRAG, lane);
#pragma unroll
    for (int ct = 0; ct < 4; ++ct) {
      v16bf bfr = load_frag(Bf + ((size_t)kk * NT16 + ct) * FRAG, lane);
      acc[ct] = wmma_bf16(af, bfr, acc[ct]);
    }
  }

  const float* feat = (f == 0) ? f0 : (f == 1) ? f1 : f2;
  const int ksel = lane >> 4, ln = lane & 15;
#pragma unroll
  for (int ct = 0; ct < 4; ++ct) {
#pragma unroll
    for (int r = 0; r < 8; ++r) {
      int o = mb + 16 * w + r + 8 * ksel;
      int n = nb + ct * 16 + ln;
      size_t fi = ((size_t)b * C_ + o) * N_ + n;
      out[(size_t)f * (B_ * C_ * N_) + fi] = feat[fi] + acc[ct][r] + bo[o];
    }
  }
}

// ---------------------------------------------------------------------------
extern "C" void kernel_launch(void* const* d_in, const int* in_sizes, int n_in,
                              void* d_out, int out_size, void* d_ws,
                              size_t ws_size, hipStream_t stream) {
  const float* f0 = (const float*)d_in[0];
  const float* f1 = (const float*)d_in[1];
  const float* f2 = (const float*)d_in[2];
  const float* Wq = (const float*)d_in[3];
  const float* bq = (const float*)d_in[4];
  const float* Wk = (const float*)d_in[5];
  const float* bk = (const float*)d_in[6];
  const float* Wv = (const float*)d_in[7];
  const float* bv = (const float*)d_in[8];
  const float* Wo = (const float*)d_in[9];
  const float* bo = (const float*)d_in[10];
  float* out = (float*)d_out;

  const size_t xsz = (size_t)F_ * B_ * CKT * NT16 * FRAG;  // 3,538,944
  const size_t wsz = 4ull * MT16 * CKT * FRAG;             // 1,048,576
  const size_t qsz = (size_t)F_ * B_ * H_ * NT16 * 2 * FRAG;
  const size_t ksz = (size_t)B_ * H_ * KVT16 * 2 * FRAG;
  const size_t vsz = (size_t)B_ * H_ * KVK * 4 * FRAG;

  bf16* Xf = (bf16*)d_ws;
  bf16* Wfp = Xf + xsz;
  bf16* Qf = Wfp + wsz;
  bf16* Kf = Qf + qsz;
  bf16* Vf = Kf + ksz;
  bf16* Of = Vf + vsz;  // total ~35.8 MB

  convert_kernel<<<2048, 256, 0, stream>>>(f0, f1, f2, Wq, Wk, Wv, Wo, Xf, Wfp);

  proj_gemm_kernel<<<dim3(C_ / 64, N_ / 64, 3 * F_ * B_), 128, 0, stream>>>(
      Wfp, Xf, bq, bk, bv, Qf, Kf, Vf);

  attn_kernel<<<dim3(N_ / 64, H_, F_ * B_), 128, 0, stream>>>(Qf, Kf, Vf, Of);

  out_gemm_kernel<<<dim3(C_ / 64, N_ / 64, F_ * B_), 128, 0, stream>>>(
      Wfp, Of, bo, f0, f1, f2, out);
}
